// MultiHeadSelfAttentionSDPA_28424093565385
// MI455X (gfx1250) — matrix operand order repaired and checked by reading, not compiled, not run on hardware
//
#include <hip/hip_runtime.h>
#include <hip/hip_bf16.h>
#include <stddef.h>

// ---------------------------------------------------------------------------
// Types for CDNA5 WMMA (wave32): D(f32 16x16) = A(bf16 16x32) * B(bf16 32x16) + C
// ---------------------------------------------------------------------------
typedef __bf16 bf16_t;
typedef __attribute__((ext_vector_type(16))) __bf16 v16bf;
typedef __attribute__((ext_vector_type(8)))  __bf16 v8bf;
typedef __attribute__((ext_vector_type(8)))  float  v8f;

static __device__ inline bf16_t f2bf(float f) {
  unsigned int u = __builtin_bit_cast(unsigned int, f);
  u += 0x7FFFu + ((u >> 16) & 1u);            // round-to-nearest-even
  unsigned short s = (unsigned short)(u >> 16);
  return __builtin_bit_cast(bf16_t, s);
}

static __device__ inline v8f wmma_bf16(v16bf a, v16bf b, v8f c) {
  // v_wmma_f32_16x16x32_bf16
  return __builtin_amdgcn_wmma_f32_16x16x32_bf16(false, a, false, b, (short)0, c,
                                                 false, false);
}

// A-matrix fragment (16x32 bf16, MxK): lane = row (lane&15),
// K elems = contiguous run {16h..16h+15}, h = lane>>4. Row-major source.
static __device__ inline v16bf load_a(const bf16_t* __restrict__ p, int ld, int lane) {
  int row = lane & 15, half = lane >> 4;
  const bf16_t* r = p + (size_t)row * ld + half * 16;
  v8bf lo = *(const v8bf*)(r);
  v8bf hi = *(const v8bf*)(r + 8);
  v16bf o;
#pragma unroll
  for (int i = 0; i < 8; ++i) { o[i] = lo[i]; o[8 + i] = hi[i]; }
  return o;
}

// B-matrix fragment (32x16 bf16, KxN): lane = column (lane&15),
// K elems = contiguous run {16h..16h+15}.  Source rows hold K contiguously
// (weight rows W[n][k], or Q rows for S^T).
static __device__ inline v16bf load_b(const bf16_t* __restrict__ p, int ld, int lane) {
  int row = lane & 15, half = lane >> 4;
  const bf16_t* r = p + (size_t)row * ld + half * 16;
  v8bf lo = *(const v8bf*)(r);
  v8bf hi = *(const v8bf*)(r + 8);
  v16bf o;
#pragma unroll
  for (int i = 0; i < 8; ++i) { o[i] = lo[i]; o[8 + i] = hi[i]; }
  return o;
}

// ---------------------------------------------------------------------------
// Kernel 1: RMSNorm (f32 in) -> bf16 activations.  One block per row.
// ---------------------------------------------------------------------------
__global__ void rmsnorm_bf16_kernel(const float* __restrict__ x,
                                    const float* __restrict__ w,
                                    bf16_t* __restrict__ xn) {
  __shared__ float red[256];
  const int row = blockIdx.x;                 // 0..4095
  const float* xr = x + (size_t)row * 2048;
  float ss = 0.f;
  for (int i = threadIdx.x; i < 2048; i += 256) { float v = xr[i]; ss += v * v; }
  red[threadIdx.x] = ss;
  __syncthreads();
  for (int s = 128; s > 0; s >>= 1) {
    if ((int)threadIdx.x < s) red[threadIdx.x] += red[threadIdx.x + s];
    __syncthreads();
  }
  const float inv = rsqrtf(red[0] * (1.0f / 2048.0f) + 1e-6f);
  bf16_t* dr = xn + (size_t)row * 2048;
  for (int i = threadIdx.x; i < 2048; i += 256) dr[i] = f2bf(xr[i] * inv * w[i]);
}

// ---------------------------------------------------------------------------
// Kernel 2: f32 -> bf16 weight conversion
// ---------------------------------------------------------------------------
__global__ void cvt_bf16_kernel(const float* __restrict__ s, bf16_t* __restrict__ d,
                                int n) {
  int i = blockIdx.x * blockDim.x + threadIdx.x;
  if (i < n) d[i] = f2bf(s[i]);
}

// ---------------------------------------------------------------------------
// Kernel 3: QKV GEMM  qkv[4096,6144] = xn[4096,2048] @ Wqkv^T, scattered into
//   q,k : [bh, t, dh]   (rows contiguous along dh -> A/B frag friendly)
//   vt  : [bh, dh, t]   (V transposed so PV^T A-frags are contiguous)
// One wave computes a 16x64 tile.  Software-pipelined (double-buffered) so the
// WMMAs of step i overlap the 10 fragment loads of step i+1.
// ---------------------------------------------------------------------------
__global__ void qkv_gemm_kernel(const bf16_t* __restrict__ xn,
                                const bf16_t* __restrict__ w,
                                bf16_t* __restrict__ q,
                                bf16_t* __restrict__ kk,
                                bf16_t* __restrict__ vt) {
  const int wave = blockIdx.x * (blockDim.x >> 5) + (threadIdx.x >> 5);
  const int lane = threadIdx.x & 31;
  const int mt = wave & 255;                  // 256 M tiles (4096/16)
  const int nt = wave >> 8;                   // 96  N tiles (6144/64)
  const int rows = mt * 16, cols = nt * 64;
  const bf16_t* A = xn + (size_t)rows * 2048;

  v8f acc[4] = {};
  // prologue: fragments for k0 = 0
  v16bf a_c = load_a(A, 2048, lane);
  v16bf b_c[4];
#pragma unroll
  for (int j = 0; j < 4; ++j)
    b_c[j] = load_b(w + (size_t)(cols + j * 16) * 2048, 2048, lane);

  for (int k0 = 32; k0 < 2048; k0 += 32) {
    // issue next step's loads first (stay outstanding across the WMMAs below)
    v16bf a_n = load_a(A + k0, 2048, lane);
    v16bf b_n[4];
#pragma unroll
    for (int j = 0; j < 4; ++j)
      b_n[j] = load_b(w + (size_t)(cols + j * 16) * 2048 + k0, 2048, lane);
    __builtin_prefetch(A + k0 + 96, 0, 1);    // global_prefetch_b8
#pragma unroll
    for (int j = 0; j < 4; ++j) acc[j] = wmma_bf16(a_c, b_c[j], acc[j]);
    a_c = a_n;
#pragma unroll
    for (int j = 0; j < 4; ++j) b_c[j] = b_n[j];
  }
#pragma unroll
  for (int j = 0; j < 4; ++j) acc[j] = wmma_bf16(a_c, b_c[j], acc[j]);

  // Epilogue: C/D layout -> lane: n = lane&15, reg r: m = r + 8*(lane>>4)
  const int half = lane >> 4, n16 = lane & 15;
#pragma unroll
  for (int j = 0; j < 4; ++j) {
    const int e = cols + j * 16 + n16;        // 0..6143
    const int which = e >> 11;                // 0=q 1=k 2=v
    const int em = e & 2047;
    const int h = em >> 7, dh = em & 127;
#pragma unroll
    for (int r = 0; r < 8; ++r) {
      const int tr = rows + r + 8 * half;     // 0..4095
      const int b = tr >> 11, t = tr & 2047;
      const int bh = b * 16 + h;
      const bf16_t v = f2bf(acc[j][r]);
      if (which == 0)      q [((size_t)bh * 2048 + t) * 128 + dh] = v;
      else if (which == 1) kk[((size_t)bh * 2048 + t) * 128 + dh] = v;
      else                 vt[((size_t)bh * 128 + dh) * 2048 + t] = v;
    }
  }
}

// ---------------------------------------------------------------------------
// Kernel 4: causal flash attention, one wave per (bh, 16-query tile).
//   S^T = K . Q^T           (A = K rows, B = Q rows)   -> softmax axis = regs
//   O^T += V^T . P^T        (A = Vt rows, B = repacked exp(S^T))
// K fragments batch-loaded before the S WMMAs; V fragments issued before the
// softmax VALU section so exp/reduce work hides their latency.
// ---------------------------------------------------------------------------
__global__ void attn_kernel(const bf16_t* __restrict__ q,
                            const bf16_t* __restrict__ k,
                            const bf16_t* __restrict__ vt,
                            bf16_t* __restrict__ y) {
  const int wave = blockIdx.x * (blockDim.x >> 5) + (threadIdx.x >> 5);
  const int lane = threadIdx.x & 31;
  const int bh = wave >> 7;                   // 0..31
  const int qt = wave & 127;
  const int qbase = qt << 4;
  const int half = lane >> 4, qn = lane & 15;

  const bf16_t* Q = q  + ((size_t)bh * 2048 + qbase) * 128;
  const bf16_t* K = k  + (size_t)bh * 2048 * 128;
  const bf16_t* V = vt + (size_t)bh * 128 * 2048;

  v16bf qf[4];
#pragma unroll
  for (int dt = 0; dt < 4; ++dt) qf[dt] = load_b(Q + dt * 32, 128, lane);

  v8f o[8] = {};
  float m_run = -1e30f, l_run = 0.f;
  const float sc = 0.08838834764831845f;      // 1/sqrt(128)
  const int nkb = (qbase + 47) >> 5;          // 32-key blocks covering causal span

  for (int jb = 0; jb < nkb; ++jb) {
    const int kb = jb << 5;
    // batch all K fragments for this 32-key block, then the 8 S^T WMMAs
    v16bf kf[8];
#pragma unroll
    for (int dt = 0; dt < 4; ++dt) {
      kf[dt]     = load_a(K + (size_t)(kb)      * 128 + dt * 32, 128, lane);
      kf[4 + dt] = load_a(K + (size_t)(kb + 16) * 128 + dt * 32, 128, lane);
    }
    v8f s0 = {}, s1 = {};
#pragma unroll
    for (int dt = 0; dt < 4; ++dt) {
      s0 = wmma_bf16(kf[dt],     qf[dt], s0);
      s1 = wmma_bf16(kf[4 + dt], qf[dt], s1);
    }
    // issue all V fragments now; the softmax below hides their latency
    v16bf vf[8];
#pragma unroll
    for (int dt = 0; dt < 8; ++dt)
      vf[dt] = load_a(V + (size_t)(dt * 16) * 2048 + kb, 2048, lane);

    // scale + additive -10000 causal mask (as in the reference)
    const int qi = qbase + qn;
    float p0[8], p1[8];
    float mb = -1e30f;
#pragma unroll
    for (int r = 0; r < 8; ++r) {
      const int key0 = kb + r + 8 * half;     // S^T rows are keys
      float v0 = s0[r] * sc + ((key0      <= qi) ? 0.f : -10000.f);
      float v1 = s1[r] * sc + ((key0 + 16 <= qi) ? 0.f : -10000.f);
      p0[r] = v0; p1[r] = v1;
      mb = fmaxf(mb, fmaxf(v0, v1));
    }
    mb = fmaxf(mb, __shfl_xor(mb, 16, 32));   // combine key halves
    const float m_new = fmaxf(m_run, mb);
    const float scal = __expf(m_run - m_new);
    float ls = 0.f;
#pragma unroll
    for (int r = 0; r < 8; ++r) {
      p0[r] = __expf(p0[r] - m_new);
      p1[r] = __expf(p1[r] - m_new);
      ls += p0[r] + p1[r];
    }
    ls += __shfl_xor(ls, 16, 32);
    l_run = l_run * scal + ls;
    m_run = m_new;

    // Repack exp(S^T) registers into the B-fragment of P^T (K = keys):
    // B-frag wants a contiguous 16-key run per half; swap the mismatched 8-key
    // sub-runs with the partner lane (lane ^ 16).
    v16bf pf;
#pragma unroll
    for (int r = 0; r < 8; ++r) {
      const float own = half ? p1[r] : p0[r];
      const float snd = half ? p0[r] : p1[r];
      const float rcv = __shfl_xor(snd, 16, 32);
      pf[r]     = f2bf(half ? rcv : own);     // keys 16h .. 16h+7
      pf[8 + r] = f2bf(half ? own : rcv);     // keys 16h+8 .. 16h+15
    }
    // O^T accumulate over 8 d-tiles (Dh=128)
#pragma unroll
    for (int dt = 0; dt < 8; ++dt) {
#pragma unroll
      for (int r = 0; r < 8; ++r) o[dt][r] = o[dt][r] * scal;
      o[dt] = wmma_bf16(vf[dt], pf, o[dt]);
    }
  }

  const float inv = 1.f / l_run;
  const int b = bh >> 4, h = bh & 15;
  bf16_t* yr = y + ((size_t)b * 2048 + qbase + qn) * 2048 + h * 128;
#pragma unroll
  for (int dt = 0; dt < 8; ++dt)
#pragma unroll
    for (int r = 0; r < 8; ++r)
      yr[dt * 16 + r + 8 * half] = f2bf(o[dt][r] * inv);
}

// ---------------------------------------------------------------------------
// Kernel 5: output projection  out[4096,2048](f32) = y @ Wout^T
// Same double-buffered pipeline as the QKV GEMM.
// ---------------------------------------------------------------------------
__global__ void out_gemm_kernel(const bf16_t* __restrict__ y,
                                const bf16_t* __restrict__ w,
                                float* __restrict__ out) {
  const int wave = blockIdx.x * (blockDim.x >> 5) + (threadIdx.x >> 5);
  const int lane = threadIdx.x & 31;
  const int mt = wave & 255;                  // 256 M tiles
  const int nt = wave >> 8;                   // 32  N tiles (2048/64)
  const int rows = mt * 16, cols = nt * 64;
  const bf16_t* A = y + (size_t)rows * 2048;

  v8f acc[4] = {};
  v16bf a_c = load_a(A, 2048, lane);
  v16bf b_c[4];
#pragma unroll
  for (int j = 0; j < 4; ++j)
    b_c[j] = load_b(w + (size_t)(cols + j * 16) * 2048, 2048, lane);

  for (int k0 = 32; k0 < 2048; k0 += 32) {
    v16bf a_n = load_a(A + k0, 2048, lane);
    v16bf b_n[4];
#pragma unroll
    for (int j = 0; j < 4; ++j)
      b_n[j] = load_b(w + (size_t)(cols + j * 16) * 2048 + k0, 2048, lane);
    __builtin_prefetch(A + k0 + 96, 0, 1);
#pragma unroll
    for (int j = 0; j < 4; ++j) acc[j] = wmma_bf16(a_c, b_c[j], acc[j]);
    a_c = a_n;
#pragma unroll
    for (int j = 0; j < 4; ++j) b_c[j] = b_n[j];
  }
#pragma unroll
  for (int j = 0; j < 4; ++j) acc[j] = wmma_bf16(a_c, b_c[j], acc[j]);

  const int half = lane >> 4, n16 = lane & 15;
#pragma unroll
  for (int j = 0; j < 4; ++j)
#pragma unroll
    for (int r = 0; r < 8; ++r) {
      const int tr = rows + r + 8 * half;
      out[(size_t)tr * 2048 + cols + j * 16 + n16] = acc[j][r];
    }
}

// ---------------------------------------------------------------------------
// Launch
// ---------------------------------------------------------------------------
extern "C" void kernel_launch(void* const* d_in, const int* in_sizes, int n_in,
                              void* d_out, int out_size, void* d_ws, size_t ws_size,
                              hipStream_t stream) {
  (void)in_sizes; (void)n_in; (void)out_size; (void)ws_size;
  const float* x    = (const float*)d_in[0];   // [2,2048,2048]
  const float* nw   = (const float*)d_in[1];   // [2048]
  const float* wqkv = (const float*)d_in[2];   // [6144,2048]
  const float* wout = (const float*)d_in[3];   // [2048,2048]
  float* out = (float*)d_out;                  // [2,2048,2048] f32

  char* ws = (char*)d_ws;
  size_t off = 0;
  auto take = [&](size_t bytes) -> void* {
    void* p = ws + off;
    off += (bytes + 255) & ~(size_t)255;
    return p;
  };
  bf16_t* xn   = (bf16_t*)take((size_t)4096 * 2048 * 2);        // 16 MB
  bf16_t* wq_b = (bf16_t*)take((size_t)6144 * 2048 * 2);        // 24 MB
  bf16_t* wo_b = (bf16_t*)take((size_t)2048 * 2048 * 2);        //  8 MB
  bf16_t* qb   = (bf16_t*)take((size_t)32 * 2048 * 128 * 2);    // 16 MB
  bf16_t* kb   = (bf16_t*)take((size_t)32 * 2048 * 128 * 2);    // 16 MB
  bf16_t* vtb  = (bf16_t*)take((size_t)32 * 128 * 2048 * 2);    // 16 MB
  bf16_t* yb   = (bf16_t*)take((size_t)4096 * 2048 * 2);        // 16 MB

  rmsnorm_bf16_kernel<<<4096, 256, 0, stream>>>(x, nw, xn);
  cvt_bf16_kernel<<<(6144 * 2048 + 255) / 256, 256, 0, stream>>>(wqkv, wq_b, 6144 * 2048);
  cvt_bf16_kernel<<<(2048 * 2048 + 255) / 256, 256, 0, stream>>>(wout, wo_b, 2048 * 2048);
  // 24576 wave-tiles (256 M x 96 N), 8 waves/block
  qkv_gemm_kernel<<<3072, 256, 0, stream>>>(xn, wq_b, qb, kb, vtb);
  // 4096 waves (32 bh x 128 q-tiles), 8 waves/block
  attn_kernel<<<512, 256, 0, stream>>>(qb, kb, vtb, yb);
  // 8192 wave-tiles (256 M x 32 N), 8 waves/block
  out_gemm_kernel<<<1024, 256, 0, stream>>>(yb, wo_b, out);
}
